// HeightAwarePointNetTiny_7902739825272
// MI455X (gfx1250) — compile-verified, matrix operand order checked by reading, not run on hardware
//
#include <hip/hip_runtime.h>
#include <hip/hip_bf16.h>
#include <hip/hip_fp16.h>

typedef __attribute__((ext_vector_type(16))) _Float16 v16h;
typedef __attribute__((ext_vector_type(8)))  _Float16 v8h;
typedef __attribute__((ext_vector_type(8)))  float    v8f;
typedef _Float16 half_t;

#define BB 4
#define NN 8192
#define NPTS (BB*NN)      // 32768 points total
#define KNB 16            // neighbors
#define K1DIM 96          // 67 padded to 3*32
#define K2DIM 160         // 131 padded to 5*32
#define K3DIM 256         // 8*32
#define NC1 256           // [u(128) | q(128)]
#define NC2 512           // [u(256) | q(256)]
#define NC3 256

// ---------------------------------------------------------------------------
// Async global->LDS copy (CDNA5, ASYNCcnt-tracked) + wait helper
// ---------------------------------------------------------------------------
__device__ __forceinline__ void async_copy16(uint32_t lds, const float4* g) {
  asm volatile("global_load_async_to_lds_b128 %0, %1, off"
               :: "v"(lds), "v"(g) : "memory");
}
__device__ __forceinline__ void wait_async0() {
  asm volatile("s_wait_asynccnt 0x0" ::: "memory");
}

// ---------------------------------------------------------------------------
// Weight repack into WMMA-B fragment order.
//   lane l: n = l&15, khalf = l>>4
//   half h in 0..7  -> K = khalf*8 + h
//   half h in 8..15 -> K = 16 + khalf*8 + (h-8)
// linear = ((ct*KSTEPS + ks)*32 + lane)*16 + h
// ---------------------------------------------------------------------------
__global__ __launch_bounds__(256) void pack_kernel(
    const float* __restrict__ b1w, const float* __restrict__ b2w,
    const float* __restrict__ h1w,
    half_t* __restrict__ p1, half_t* __restrict__ p2, half_t* __restrict__ p3) {
  int id = blockIdx.x * 256 + threadIdx.x;
  // ---- pack1: 16 coltiles x 3 ksteps x 32 lanes x 16 halves = 24576
  if (id < 24576) {
    int h = id & 15, lane = (id >> 4) & 31;
    int comb = id >> 9, ks = comb % 3, ct = comb / 3;
    int n = lane & 15, kh = lane >> 4;
    int kk = ks * 32 + ((h < 8) ? (kh * 8 + h) : (16 + kh * 8 + (h - 8)));
    int col = ct * 16 + n;
    float v = 0.f;
    if (col < 128) {                 // u-weights: Wb rows 64..127, Wc rows 128..130
      if (kk < 64)      v = b1w[(64 + kk) * 128 + col];
      else if (kk < 67) v = b1w[(128 + kk - 64) * 128 + col];
    } else {                         // q-weights: Wa - Wb, -Wc
      int c = col - 128;
      if (kk < 64)      v = b1w[kk * 128 + c] - b1w[(64 + kk) * 128 + c];
      else if (kk < 67) v = -b1w[(128 + kk - 64) * 128 + c];
    }
    p1[id] = (half_t)v;
    return;
  }
  id -= 24576;
  // ---- pack2: 32 coltiles x 5 ksteps x 512 = 81920
  if (id < 81920) {
    int h = id & 15, lane = (id >> 4) & 31;
    int comb = id >> 9, ks = comb % 5, ct = comb / 5;
    int n = lane & 15, kh = lane >> 4;
    int kk = ks * 32 + ((h < 8) ? (kh * 8 + h) : (16 + kh * 8 + (h - 8)));
    int col = ct * 16 + n;
    float v = 0.f;
    if (col < 256) {
      if (kk < 128)      v = b2w[(128 + kk) * 256 + col];
      else if (kk < 131) v = b2w[(256 + kk - 128) * 256 + col];
    } else {
      int c = col - 256;
      if (kk < 128)      v = b2w[kk * 256 + c] - b2w[(128 + kk) * 256 + c];
      else if (kk < 131) v = -b2w[(256 + kk - 128) * 256 + c];
    }
    p2[id] = (half_t)v;
    return;
  }
  id -= 81920;
  // ---- pack3: head1 f-part (rows 0..255): 16 coltiles x 8 ksteps x 512 = 65536
  if (id < 65536) {
    int h = id & 15, lane = (id >> 4) & 31;
    int comb = id >> 9, ks = comb & 7, ct = comb >> 3;
    int n = lane & 15, kh = lane >> 4;
    int kk = ks * 32 + ((h < 8) ? (kh * 8 + h) : (16 + kh * 8 + (h - 8)));
    p3[id] = (half_t)h1w[kk * 256 + (ct * 16 + n)];
  }
}

// ---------------------------------------------------------------------------
// Stem: coords + relu(x @ stem_w + stem_b); writes A1 row [f0(64)|p(3)|pad]
// ---------------------------------------------------------------------------
__global__ __launch_bounds__(256) void stem_kernel(
    const float* __restrict__ x,
    const float* __restrict__ ha, const float* __restrict__ hb, const float* __restrict__ hc,
    const float* __restrict__ sw, const float* __restrict__ sb,
    float4* __restrict__ coords, half_t* __restrict__ A1) {
  __shared__ float w[256];
  __shared__ float bsh[64];
  int t = threadIdx.x;
  w[t] = sw[t];
  if (t < 64) bsh[t] = sb[t];
  __syncthreads();
  int i = blockIdx.x * 256 + t;
  float4 xi = ((const float4*)x)[i];
  float z = (*ha) * xi.z + (*hb) * xi.w + (*hc);
  coords[i] = make_float4(xi.x, xi.y, z, 0.f);
  half_t* row = A1 + (size_t)i * K1DIM;
#pragma unroll
  for (int oc = 0; oc < 64; ++oc) {
    float v = bsh[oc] + xi.x * w[oc] + xi.y * w[64 + oc] + xi.z * w[128 + oc] + xi.w * w[192 + oc];
    row[oc] = (half_t)fmaxf(v, 0.f);
  }
  row[64] = (half_t)xi.x; row[65] = (half_t)xi.y; row[66] = (half_t)z;
#pragma unroll
  for (int j = 67; j < K1DIM; ++j) row[j] = (half_t)0.f;
}

// ---------------------------------------------------------------------------
// KNN: thread-per-query, double-buffered ASYNC global->LDS candidate tiles,
// register top-16 (replace-worst policy).
// ---------------------------------------------------------------------------
__global__ __launch_bounds__(256) void knn_kernel(
    const float4* __restrict__ coords, int* __restrict__ idxbuf) {
  __shared__ float4 tile[2][256];
  int t = threadIdx.x;
  int b = blockIdx.x >> 5;                     // 32 blocks per batch
  int qi = ((blockIdx.x & 31) << 8) + t;
  const float4* src = coords + (size_t)b * NN;
  float4 q = src[qi];
  float bd[KNB]; int bi[KNB];
#pragma unroll
  for (int j = 0; j < KNB; ++j) { bd[j] = 3.4e38f; bi[j] = 0; }
  float worst = 3.4e38f; int wslot = 0;

  uint32_t lds0 = (uint32_t)(size_t)(&tile[0][t]);
  uint32_t lds1 = (uint32_t)(size_t)(&tile[1][t]);

  // prime buffer 0
  async_copy16(lds0, src + t);
  wait_async0();
  __syncthreads();

  int buf = 0;
  for (int t0 = 0; t0 < NN; t0 += 256) {
    if (t0 + 256 < NN)                          // prefetch next tile
      async_copy16(buf ? lds0 : lds1, src + t0 + 256 + t);
    const float4* cur = tile[buf];
    for (int s = 0; s < 256; ++s) {
      float4 p = cur[s];
      float dx = q.x - p.x, dy = q.y - p.y, dz = q.z - p.z;
      float d = dx * dx + dy * dy + dz * dz;
      if (d < worst) {
#pragma unroll
        for (int j = 0; j < KNB; ++j) if (j == wslot) { bd[j] = d; bi[j] = t0 + s; }
        worst = bd[0]; wslot = 0;
#pragma unroll
        for (int j = 1; j < KNB; ++j) if (bd[j] > worst) { worst = bd[j]; wslot = j; }
      }
    }
    wait_async0();
    __syncthreads();
    buf ^= 1;
  }
  int* op = idxbuf + (size_t)(b * NN + qi) * KNB;
#pragma unroll
  for (int j = 0; j < KNB; ++j) op[j] = bi[j];
}

// ---------------------------------------------------------------------------
// WMMA GEMM: C[32768 x ncols] = A[32768 x kdim] * Wpack.
// One wave computes a 16-row x 64-col strip (4 tiles): one A-fragment load
// feeds 4 WMMAs per k-step (4x less A traffic than tile-per-wave).
// Optional per-batch row bias + relu on store.
// ---------------------------------------------------------------------------
__global__ __launch_bounds__(256) void gemm_kernel(
    const half_t* __restrict__ A, int kdim,
    const half_t* __restrict__ Wp, int ksteps,
    float* __restrict__ C, int ncols,
    const float* __restrict__ rowBias, int doRelu) {
  int lane = threadIdx.x & 31;
  int wave = threadIdx.x >> 5;
  int tid = blockIdx.x * 8 + wave;
  int groupsx = ncols >> 6;                 // 64-col groups
  int rt = tid / groupsx, cg = tid - rt * groupsx;
  int m = lane & 15, kh = lane >> 4;
  const half_t* ap = A + (size_t)(rt * 16 + m) * kdim + kh * 8;
  const size_t tstride = (size_t)ksteps * 512;        // halves per col tile
  const half_t* wbase = Wp + (size_t)(cg * 4) * tstride + (size_t)lane * 16;
  v8f acc0 = {}, acc1 = {}, acc2 = {}, acc3 = {};
  for (int ks = 0; ks < ksteps; ++ks) {
    union { v16h v; v8h h[2]; } a;
    a.h[0] = *(const v8h*)(ap);
    a.h[1] = *(const v8h*)(ap + 16);
    size_t ko = (size_t)ks * 512;
    v16h w0 = *(const v16h*)(wbase + 0 * tstride + ko);
    v16h w1 = *(const v16h*)(wbase + 1 * tstride + ko);
    v16h w2 = *(const v16h*)(wbase + 2 * tstride + ko);
    v16h w3 = *(const v16h*)(wbase + 3 * tstride + ko);
    acc0 = __builtin_amdgcn_wmma_f32_16x16x32_f16(false, a.v, false, w0, (short)0, acc0, false, false);
    acc1 = __builtin_amdgcn_wmma_f32_16x16x32_f16(false, a.v, false, w1, (short)0, acc1, false, false);
    acc2 = __builtin_amdgcn_wmma_f32_16x16x32_f16(false, a.v, false, w2, (short)0, acc2, false, false);
    acc3 = __builtin_amdgcn_wmma_f32_16x16x32_f16(false, a.v, false, w3, (short)0, acc3, false, false);
    ap += 32;
  }
  int row0 = rt * 16;
  int batch = row0 >> 13;
  v8f accs[4] = {acc0, acc1, acc2, acc3};
#pragma unroll
  for (int j = 0; j < 4; ++j) {
    int col = (cg * 4 + j) * 16 + m;
    float rb = 0.f;
    if (rowBias) rb = rowBias[batch * ncols + col];
#pragma unroll
    for (int r = 0; r < 8; ++r) {
      float v = accs[j][r] + rb;
      if (doRelu) v = fmaxf(v, 0.f);
      C[(size_t)(row0 + r + 8 * kh) * ncols + col] = v;
    }
  }
}

// ---------------------------------------------------------------------------
// combine1: f1[i][c] = max_k relu(q_i[c] + u_{nb_k}[c] + b1_b[c]); writes A2
// ---------------------------------------------------------------------------
__global__ __launch_bounds__(256) void combine1_kernel(
    const float* __restrict__ out1, const int* __restrict__ idxbuf,
    const float* __restrict__ b1b, const float4* __restrict__ coords,
    half_t* __restrict__ A2) {
  int c = threadIdx.x;                       // 0..127
  int i = blockIdx.x * 2 + threadIdx.y;
  int b = i >> 13;
  const int* nb = idxbuf + (size_t)i * KNB;
  float q = out1[(size_t)i * NC1 + 128 + c] + b1b[c];
  float acc = 0.f;
#pragma unroll 4
  for (int k = 0; k < KNB; ++k) {
    int j = nb[k];
    float u = out1[(size_t)(b * NN + j) * NC1 + c];
    acc = fmaxf(acc, fmaxf(q + u, 0.f));
  }
  half_t* row = A2 + (size_t)i * K2DIM;
  row[c] = (half_t)acc;
  if (c < 32) {
    float4 p = coords[i];
    float v = (c == 0) ? p.x : (c == 1) ? p.y : (c == 2) ? p.z : 0.f;
    row[128 + c] = (half_t)v;
  }
}

// combine2: f2[i][c] = max_k relu(q2 + u2 + b2_b); writes A3 (f16, 256 wide)
__global__ __launch_bounds__(256) void combine2_kernel(
    const float* __restrict__ out2, const int* __restrict__ idxbuf,
    const float* __restrict__ b2b, half_t* __restrict__ A3) {
  int c = threadIdx.x;                       // 0..255
  int i = blockIdx.x;
  int b = i >> 13;
  const int* nb = idxbuf + (size_t)i * KNB;
  float q = out2[(size_t)i * NC2 + 256 + c] + b2b[c];
  float acc = 0.f;
#pragma unroll 4
  for (int k = 0; k < KNB; ++k) {
    int j = nb[k];
    float u = out2[(size_t)(b * NN + j) * NC2 + c];
    acc = fmaxf(acc, fmaxf(q + u, 0.f));
  }
  A3[(size_t)i * K3DIM + c] = (half_t)acc;
}

// Per-batch channel max over 256-point chunks -> partial[4*32][256]
__global__ __launch_bounds__(256) void reduce_kernel(
    const half_t* __restrict__ A3, float* __restrict__ partial) {
  int c = threadIdx.x;
  int b = blockIdx.x >> 5, chunk = blockIdx.x & 31;
  int base = b * NN + chunk * 256;
  float m = 0.f;
  for (int n = 0; n < 256; ++n)
    m = fmaxf(m, (float)A3[(size_t)(base + n) * K3DIM + c]);
  partial[blockIdx.x * 256 + c] = m;
}

// Global MLP: gmax -> g=relu(gmax@glob_w+glob_b); gb = g@head1_w[256:512] + h1b
__global__ __launch_bounds__(256) void ghead_kernel(
    const float* __restrict__ partial,
    const float* __restrict__ gw, const float* __restrict__ gbias,
    const float* __restrict__ h1w, const float* __restrict__ h1b,
    float* __restrict__ gb) {
  __shared__ float gm[BB][256];
  __shared__ float gg[BB][256];
  int c = threadIdx.x;
  for (int b = 0; b < BB; ++b) {
    float m = 0.f;
    for (int s = 0; s < 32; ++s) m = fmaxf(m, partial[(b * 32 + s) * 256 + c]);
    gm[b][c] = m;
  }
  __syncthreads();
  for (int b = 0; b < BB; ++b) {
    float acc = gbias[c];
    for (int k = 0; k < 256; ++k) acc += gm[b][k] * gw[k * 256 + c];
    gg[b][c] = fmaxf(acc, 0.f);
  }
  __syncthreads();
  for (int b = 0; b < BB; ++b) {
    float acc = h1b[c];
    for (int k = 0; k < 256; ++k) acc += gg[b][k] * h1w[(256 + k) * 256 + c];
    gb[b * 256 + c] = acc;
  }
}

// Final: logits = h @ head2_w + head2_b, class-0 height bias, write d_out
__global__ __launch_bounds__(256) void final_kernel(
    const float* __restrict__ h, const float* __restrict__ x,
    const float* __restrict__ h2w, const float* __restrict__ h2b,
    const float* __restrict__ thresh, const float* __restrict__ sharp,
    const float* __restrict__ scale, float* __restrict__ out) {
  __shared__ float w[768];
  __shared__ float bsh[3];
  int t = threadIdx.x;
  for (int j = t; j < 768; j += 256) w[j] = h2w[j];
  if (t < 3) bsh[t] = h2b[t];
  __syncthreads();
  int i = blockIdx.x * 256 + t;
  const float* hr = h + (size_t)i * 256;
  float l0 = bsh[0], l1 = bsh[1], l2 = bsh[2];
  for (int k = 0; k < 256; ++k) {
    float v = hr[k];
    l0 += v * w[k * 3 + 0];
    l1 += v * w[k * 3 + 1];
    l2 += v * w[k * 3 + 2];
  }
  float hag = x[(size_t)i * 4 + 3];
  float sig = 1.f / (1.f + __expf(-(*sharp) * ((*thresh) - hag)));
  l0 += (*scale) * sig;
  float* o = out + (size_t)i * 3;
  o[0] = l0; o[1] = l1; o[2] = l2;
}

// ---------------------------------------------------------------------------
static size_t alignup(size_t v) { return (v + 255) & ~(size_t)255; }

extern "C" void kernel_launch(void* const* d_in, const int* in_sizes, int n_in,
                              void* d_out, int out_size, void* d_ws, size_t ws_size,
                              hipStream_t stream) {
  (void)in_sizes; (void)n_in; (void)out_size; (void)ws_size;
  const float* x      = (const float*)d_in[0];
  const float* hmix_a = (const float*)d_in[1];
  const float* hmix_b = (const float*)d_in[2];
  const float* hmix_c = (const float*)d_in[3];
  const float* stem_w = (const float*)d_in[4];
  const float* stem_b = (const float*)d_in[5];
  const float* b1_w   = (const float*)d_in[6];
  const float* b1_b   = (const float*)d_in[7];
  const float* b2_w   = (const float*)d_in[8];
  const float* b2_b   = (const float*)d_in[9];
  const float* glob_w = (const float*)d_in[10];
  const float* glob_b = (const float*)d_in[11];
  const float* head1_w= (const float*)d_in[12];
  const float* head1_b= (const float*)d_in[13];
  const float* head2_w= (const float*)d_in[14];
  const float* head2_b= (const float*)d_in[15];
  const float* thresh = (const float*)d_in[16];
  const float* sharp  = (const float*)d_in[17];
  const float* scale  = (const float*)d_in[18];
  float* out = (float*)d_out;

  char* ws = (char*)d_ws;
  size_t off = 0;
  float4* coords = (float4*)(ws + off); off = alignup(off + (size_t)NPTS * 16);
  half_t* A1     = (half_t*)(ws + off); off = alignup(off + (size_t)NPTS * K1DIM * 2);
  half_t* A2     = (half_t*)(ws + off); off = alignup(off + (size_t)NPTS * K2DIM * 2);
  half_t* A3     = (half_t*)(ws + off); off = alignup(off + (size_t)NPTS * K3DIM * 2);
  int*    idxb   = (int*)(ws + off);    off = alignup(off + (size_t)NPTS * KNB * 4);
  float*  partial= (float*)(ws + off);  off = alignup(off + (size_t)BB * 32 * 256 * 4);
  float*  gb     = (float*)(ws + off);  off = alignup(off + (size_t)BB * 256 * 4);
  half_t* p1     = (half_t*)(ws + off); off = alignup(off + (size_t)24576 * 2);
  half_t* p2     = (half_t*)(ws + off); off = alignup(off + (size_t)81920 * 2);
  half_t* p3     = (half_t*)(ws + off); off = alignup(off + (size_t)65536 * 2);
  float*  gout   = (float*)(ws + off);  // 32768*512 f32 (reused: out1, out2, h)

  // 1) repack weights into WMMA fragment order (172032 elements)
  pack_kernel<<<672, 256, 0, stream>>>(b1_w, b2_w, head1_w, p1, p2, p3);
  // 2) stem + coords
  stem_kernel<<<NPTS / 256, 256, 0, stream>>>(x, hmix_a, hmix_b, hmix_c,
                                              stem_w, stem_b, coords, A1);
  // 3) knn (async double-buffered candidate staging)
  knn_kernel<<<NPTS / 256, 256, 0, stream>>>(coords, idxb);
  // 4) layer1 GEMM: [u|q] = A1 @ pack1  (2048 rowtiles x 4 colgroups)
  gemm_kernel<<<(2048 * (NC1 / 64)) / 8, 256, 0, stream>>>(
      A1, K1DIM, p1, K1DIM / 32, gout, NC1, nullptr, 0);
  // 5) combine1 -> A2
  combine1_kernel<<<NPTS / 2, dim3(128, 2), 0, stream>>>(gout, idxb, b1_b, coords, A2);
  // 6) layer2 GEMM: [u2|q2] = A2 @ pack2 (2048 x 8 colgroups)
  gemm_kernel<<<(2048 * (NC2 / 64)) / 8, 256, 0, stream>>>(
      A2, K2DIM, p2, K2DIM / 32, gout, NC2, nullptr, 0);
  // 7) combine2 -> A3 (f2)
  combine2_kernel<<<NPTS, 256, 0, stream>>>(gout, idxb, b2_b, A3);
  // 8) per-batch max-pool partials
  reduce_kernel<<<BB * 32, 256, 0, stream>>>(A3, partial);
  // 9) global MLP -> per-batch bias gb
  ghead_kernel<<<1, 256, 0, stream>>>(partial, glob_w, glob_b, head1_w, head1_b, gb);
  // 10) head1 GEMM: h = relu(f2 @ W1a + gb[batch])  (reuses gout as h)
  gemm_kernel<<<(2048 * (NC3 / 64)) / 8, 256, 0, stream>>>(
      A3, K3DIM, p3, K3DIM / 32, gout, NC3, gb, 1);
  // 11) head2 + height bias -> logits
  final_kernel<<<NPTS / 256, 256, 0, stream>>>(gout, x, head2_w, head2_b,
                                               thresh, sharp, scale, out);
}
